// GraphAttentionHeadModule_39436389711886
// MI455X (gfx1250) — matrix operand order repaired and checked by reading, not compiled, not run on hardware
//
#include <hip/hip_runtime.h>
#include <hip/hip_bf16.h>
#include <stdint.h>

typedef _Float16 v16h __attribute__((ext_vector_type(16)));
typedef float    v8f  __attribute__((ext_vector_type(8)));

#define N_NODES 8192
#define D_IN    512
#define D_OUT   256
#define ALPHA   0.2f
#define WORDS_PER_ROW 256  // 8192 / 32

__device__ __forceinline__ v8f wmma_f16(v16h a, v16h b, v8f c) {
    return __builtin_amdgcn_wmma_f32_16x16x32_f16(
        /*neg_a=*/false, a, /*neg_b=*/false, b,
        /*c_mod=*/(short)0, c, /*reuse_a=*/false, /*reuse_b=*/false);
}

// ---------------------------------------------------------------------------
// K0: pack adjacency (int32 0/1) into a bitmask: bits[i][w] covers j = 32w..32w+31
// 256 MB read once; both softmax passes then read 8 MB instead of 256 MB each.
// ---------------------------------------------------------------------------
__global__ void k_pack_bits(const int* __restrict__ adj, unsigned* __restrict__ bits) {
    long long w = (long long)blockIdx.x * 256 + threadIdx.x;   // word index, 2M total
    const int4* p = (const int4*)(adj + w * 32);
    unsigned m = 0;
#pragma unroll
    for (int q = 0; q < 8; ++q) {
        int4 v = p[q];
        m |= ((unsigned)(v.x > 0) << (4 * q + 0)) |
             ((unsigned)(v.y > 0) << (4 * q + 1)) |
             ((unsigned)(v.z > 0) << (4 * q + 2)) |
             ((unsigned)(v.w > 0) << (4 * q + 3));
    }
    bits[w] = m;
}

// ---------------------------------------------------------------------------
// K1: W_phi [512,256] f32 -> W16T [256][512] f16 (transposed, for B-operand loads)
// ---------------------------------------------------------------------------
__global__ void k_cvt_w(const float* __restrict__ W, _Float16* __restrict__ WT) {
    int idx = blockIdx.x * 256 + threadIdx.x;    // 0 .. 512*256-1
    int k = idx >> 8, c = idx & 255;
    WT[c * D_IN + k] = (_Float16)W[idx];
}

// ---------------------------------------------------------------------------
// K2: h = feat @ W_phi + b_phi via WMMA f16; writes hT (f16, [256][8192] transposed)
//     and fuses mu = h@w_mu + b_mu, xi = h@w_xi + b_xi via shfl reductions.
// Block: 128 threads = 4 waves; wave = one 16-row tile; BM = 64; grid = 128.
// B tiles software-pipelined one deep so each WMMA overlaps the next B load.
// ---------------------------------------------------------------------------
__global__ void __launch_bounds__(128, 1) __attribute__((amdgpu_waves_per_eu(1, 2)))
k_gemm_h(const float* __restrict__ feat, const _Float16* __restrict__ WT,
         const float* __restrict__ bphi,
         const float* __restrict__ wmu, const float* __restrict__ bmu,
         const float* __restrict__ wxi, const float* __restrict__ bxi,
         _Float16* __restrict__ hT, float* __restrict__ mu,
         float* __restrict__ xi) {
    const int lane = threadIdx.x & 31, wid = threadIdx.x >> 5;
    const int n16 = lane & 15, hi = lane >> 4;
    const int rowBase = blockIdx.x * 64 + wid * 16;
    const int arow = rowBase + n16;                  // A-operand row for this lane

    v8f acc[16];
#pragma unroll
    for (int t = 0; t < 16; ++t) acc[t] = (v8f){0.f,0.f,0.f,0.f,0.f,0.f,0.f,0.f};

    for (int kc = 0; kc < D_IN / 32; ++kc) {
        const int k0 = kc * 32;
        const float* ap = feat + (long long)arow * D_IN + k0;
        // lane k-set: [16*hi .. 16*hi+15]  (contiguous, matches B)
        float4 fa = *(const float4*)(ap + 16 * hi);
        float4 fb = *(const float4*)(ap + 16 * hi + 4);
        float4 fc = *(const float4*)(ap + 16 * hi + 8);
        float4 fd = *(const float4*)(ap + 16 * hi + 12);
        v16h A;
        A[0]=(_Float16)fa.x; A[1]=(_Float16)fa.y; A[2]=(_Float16)fa.z; A[3]=(_Float16)fa.w;
        A[4]=(_Float16)fb.x; A[5]=(_Float16)fb.y; A[6]=(_Float16)fb.z; A[7]=(_Float16)fb.w;
        A[8]=(_Float16)fc.x; A[9]=(_Float16)fc.y; A[10]=(_Float16)fc.z; A[11]=(_Float16)fc.w;
        A[12]=(_Float16)fd.x; A[13]=(_Float16)fd.y; A[14]=(_Float16)fd.z; A[15]=(_Float16)fd.w;

        // B(k,n) = W[k0+k][16t+n] = WT[(16t+n)*512 + k0 + k]; lane holds col n16,
        // elements i -> k = 16*hi + i (contiguous, 32B-aligned load). Pipeline 1-deep.
        const _Float16* bp = WT + (long long)n16 * D_IN + k0 + 16 * hi;
        v16h B0 = *(const v16h*)(bp);
#pragma unroll
        for (int t = 0; t < 16; ++t) {
            v16h B1;
            if (t < 15) B1 = *(const v16h*)(bp + (long long)(t + 1) * 16 * D_IN);
            acc[t] = wmma_f16(A, B0, acc[t]);
            B0 = B1;
        }
    }

    float pmu[8], pxi[8];
#pragma unroll
    for (int r = 0; r < 8; ++r) { pmu[r] = 0.f; pxi[r] = 0.f; }

#pragma unroll
    for (int t = 0; t < 16; ++t) {
        const int col = 16 * t + n16;
        const float bv = bphi[col], wm = wmu[col], wx = wxi[col];
#pragma unroll
        for (int r = 0; r < 8; ++r) {
            // D layout: lane col = n16, VGPR r -> row m = r + 8*hi
            const float h = acc[t][r] + bv;
            const int   ri = rowBase + r + 8 * hi;
            hT[(long long)col * N_NODES + ri] = (_Float16)h;
            pmu[r] += h * wm;
            pxi[r] += h * wx;
        }
    }
#pragma unroll
    for (int r = 0; r < 8; ++r) {
#pragma unroll
        for (int s = 1; s <= 8; s <<= 1) {           // reduce across 16 lanes of same hi-group
            pmu[r] += __shfl_xor(pmu[r], s, 32);
            pxi[r] += __shfl_xor(pxi[r], s, 32);
        }
        if (n16 == 0) {
            const int ri = rowBase + r + 8 * hi;
            mu[ri] = pmu[r] + bmu[0];
            xi[ri] = pxi[r] + bxi[0];
        }
    }
}

// ---------------------------------------------------------------------------
// K3: per-row softmax stats. leaky is monotone, so rowmax(e) = leaky(mu_i + max xi_j).
// Block: 256 threads handles 16 rows; xi cached in LDS (32 KB).
// ---------------------------------------------------------------------------
__global__ void k_row_stats(const unsigned* __restrict__ bits, const float* __restrict__ mu,
                            const float* __restrict__ xi, float* __restrict__ mrow,
                            float* __restrict__ zinv, int* __restrict__ deg) {
    __shared__ float xiS[N_NODES];
    __shared__ float rf[8];
    __shared__ int   ri[8];
    __shared__ float bcm;
    __shared__ int   bcc;
    const int t = threadIdx.x, lane = t & 31, wid = t >> 5;
    for (int k = t; k < N_NODES; k += 256) xiS[k] = xi[k];
    __syncthreads();

    for (int rr = 0; rr < 16; ++rr) {
        const int i = blockIdx.x * 16 + rr;
        const unsigned w = bits[(long long)i * WORDS_PER_ROW + t];
        const float mui = mu[i];
        float mx = -3.4e38f;
        int   cnt = __popc(w);
        for (int b = 0; b < 32; ++b)
            if ((w >> b) & 1u) { const float v = xiS[t * 32 + b]; mx = v > mx ? v : mx; }
#pragma unroll
        for (int s = 16; s > 0; s >>= 1) {
            const float o = __shfl_xor(mx, s, 32); mx = o > mx ? o : mx;
            cnt += __shfl_xor(cnt, s, 32);
        }
        if (lane == 0) { rf[wid] = mx; ri[wid] = cnt; }
        __syncthreads();
        if (t == 0) {
            float m2 = rf[0]; int c2 = ri[0];
            for (int q = 1; q < 8; ++q) { m2 = rf[q] > m2 ? rf[q] : m2; c2 += ri[q]; }
            const float e = mui + m2;
            bcm = e >= 0.f ? e : ALPHA * e;   // leaky of masked max
            bcc = c2;
        }
        __syncthreads();
        const float mi = bcm;
        const int ctot = bcc;
        float s = 0.f;
        if (ctot > 0) {
            for (int b = 0; b < 32; ++b)
                if ((w >> b) & 1u) {
                    const float e = mui + xiS[t * 32 + b];
                    const float l = e >= 0.f ? e : ALPHA * e;
                    s += __expf(l - mi);
                }
        }
#pragma unroll
        for (int sf = 16; sf > 0; sf >>= 1) s += __shfl_xor(s, sf, 32);
        __syncthreads();
        if (lane == 0) rf[wid] = s;
        __syncthreads();
        if (t == 0) {
            float s2 = 0.f;
            for (int q = 0; q < 8; ++q) s2 += rf[q];
            if (ctot > 0) { mrow[i] = mi; zinv[i] = 1.f / s2; }
            else          { mrow[i] = 0.f; zinv[i] = 1.f / (float)N_NODES; }  // empty row -> uniform softmax
            deg[i] = ctot;
        }
        __syncthreads();
    }
}

// ---------------------------------------------------------------------------
// K4: out = elu( softmax-weighted aggregation of h ), fully fused.
// Block: 256 threads = 8 waves; BM=64 rows; wave = (row-tile r = wid>>1, j-parity = wid&1).
// A-operand (16x32 softmax weights) built IN REGISTERS per lane (one exp per (i,j) total);
// B-operand = hT tiles via two 32B-aligned global v16h loads (L2 resident, 4 MB),
// software-pipelined one tile deep so each WMMA overlaps the next B load.
// Parities combined through LDS (stride padded to 260 to kill bank conflicts).
// ---------------------------------------------------------------------------
__global__ void __launch_bounds__(256, 1) __attribute__((amdgpu_waves_per_eu(1, 2)))
k_attn(const unsigned* __restrict__ bits, const _Float16* __restrict__ hT,
       const float* __restrict__ mu, const float* __restrict__ xi,
       const float* __restrict__ mrow, const float* __restrict__ zinv,
       const int* __restrict__ deg, float* __restrict__ out) {
    __shared__ float cmb[64 * 260];
    const int lane = threadIdx.x & 31, wid = threadIdx.x >> 5;
    const int n16 = lane & 15, hi = lane >> 4;
    const int r = wid >> 1, par = wid & 1;
    const int i0 = blockIdx.x * 64 + 16 * r;
    const int iL = i0 + n16;                    // A-operand row for this lane
    const float muL = mu[iL];
    const float mL  = mrow[iL];
    const int   degL = deg[iL];

    v8f acc[16];
#pragma unroll
    for (int t = 0; t < 16; ++t) acc[t] = (v8f){0.f,0.f,0.f,0.f,0.f,0.f,0.f,0.f};

    for (int jc = par; jc < WORDS_PER_ROW; jc += 2) {
        const int j0 = jc * 32;
        const unsigned w = bits[(long long)iL * WORDS_PER_ROW + jc];
        __builtin_prefetch(bits + (long long)iL * WORDS_PER_ROW + jc + 2, 0, 0);

        const float4* xp = (const float4*)(xi + j0);
        const float4 xa = xp[4 * hi], xb = xp[4 * hi + 1];
        const float4 xc = xp[4 * hi + 2], xd = xp[4 * hi + 3];
        float xs[16];
        xs[0]=xa.x; xs[1]=xa.y; xs[2]=xa.z; xs[3]=xa.w;
        xs[4]=xb.x; xs[5]=xb.y; xs[6]=xb.z; xs[7]=xb.w;
        xs[8]=xc.x; xs[9]=xc.y; xs[10]=xc.z; xs[11]=xc.w;
        xs[12]=xd.x; xs[13]=xd.y; xs[14]=xd.z; xs[15]=xd.w;

        v16h A;
#pragma unroll
        for (int i = 0; i < 16; ++i) {
            const int k = 16 * hi + i;                       // contiguous k index (matches B)
            const float e = muL + xs[i];
            const float l = e >= 0.f ? e : ALPHA * e;        // LeakyReLU
            float wv = ((w >> k) & 1u) ? __expf(l - mL) : 0.f;
            wv = degL > 0 ? wv : 1.0f;                       // empty row -> uniform weights
            A[i] = (_Float16)wv;
        }

        const _Float16* bp = hT + (long long)n16 * N_NODES + j0 + 16 * hi;
        v16h B0 = *(const v16h*)(bp);
#pragma unroll
        for (int t = 0; t < 16; ++t) {
            v16h B1;
            if (t < 15) B1 = *(const v16h*)(bp + (long long)(t + 1) * 16 * N_NODES);
            acc[t] = wmma_f16(A, B0, acc[t]);
            B0 = B1;
        }
    }

    if (par == 1) {
#pragma unroll
        for (int t = 0; t < 16; ++t)
#pragma unroll
            for (int rc = 0; rc < 8; ++rc)
                cmb[(16 * r + rc + 8 * hi) * 260 + 16 * t + n16] = acc[t][rc];
    }
    __syncthreads();
    if (par == 0) {
        float zv[8];
#pragma unroll
        for (int rc = 0; rc < 8; ++rc) zv[rc] = zinv[i0 + rc + 8 * hi];
#pragma unroll
        for (int t = 0; t < 16; ++t) {
#pragma unroll
            for (int rc = 0; rc < 8; ++rc) {
                float v = acc[t][rc] + cmb[(16 * r + rc + 8 * hi) * 260 + 16 * t + n16];
                v *= zv[rc];
                v = v > 0.f ? v : (__expf(v) - 1.f);         // ELU
                out[(long long)(i0 + rc + 8 * hi) * D_OUT + 16 * t + n16] = v;
            }
        }
    }
}

// ---------------------------------------------------------------------------
extern "C" void kernel_launch(void* const* d_in, const int* in_sizes, int n_in,
                              void* d_out, int out_size, void* d_ws, size_t ws_size,
                              hipStream_t stream) {
    const float* feat = (const float*)d_in[0];
    const int*   adj  = (const int*)d_in[1];
    const float* Wphi = (const float*)d_in[2];
    const float* bphi = (const float*)d_in[3];
    const float* wmu  = (const float*)d_in[4];
    const float* bmu  = (const float*)d_in[5];
    const float* wxi  = (const float*)d_in[6];
    const float* bxi  = (const float*)d_in[7];
    float* out = (float*)d_out;

    // workspace layout (~12.4 MB)
    char* ws = (char*)d_ws;
    unsigned*  bits = (unsigned*)(ws);                         // 8 MB
    _Float16*  hT   = (_Float16*)(ws + (8u << 20));            // 4 MB  [256][8192]
    _Float16*  W16T = (_Float16*)(ws + (12u << 20));           // 256 KB [256][512]
    float*     mu   = (float*)(ws + (12u << 20) + (256u << 10));         // 32 KB
    float*     xi   = (float*)(ws + (12u << 20) + (288u << 10));         // 32 KB
    float*     mrow = (float*)(ws + (12u << 20) + (320u << 10));         // 32 KB
    float*     zinv = (float*)(ws + (12u << 20) + (352u << 10));         // 32 KB
    int*       deg  = (int*)  (ws + (12u << 20) + (384u << 10));         // 32 KB

    k_pack_bits<<<N_NODES * WORDS_PER_ROW / 256, 256, 0, stream>>>(adj, bits);
    k_cvt_w<<<(D_IN * D_OUT) / 256, 256, 0, stream>>>(Wphi, W16T);
    k_gemm_h<<<N_NODES / 64, 128, 0, stream>>>(feat, W16T, bphi, wmu, bmu, wxi, bxi,
                                               hT, mu, xi);
    k_row_stats<<<N_NODES / 16, 256, 0, stream>>>(bits, mu, xi, mrow, zinv, deg);
    k_attn<<<N_NODES / 64, 256, 0, stream>>>(bits, hT, mu, xi, mrow, zinv, deg, out);
}
